// LocalAttention_16527034155052
// MI455X (gfx1250) — matrix operand order repaired and checked by reading, not compiled, not run on hardware
//
#include <hip/hip_runtime.h>

// ---------------------------------------------------------------------------
// Local (sliding-window) attention for MI455X / gfx1250, wave32 + WMMA bf16.
//
// Math trick: RoPE(q,p)·RoPE(k,p') depends only on p'-p, and here
// p_k - p_q == t_k - t_q, so we rope Q and K once at absolute position t.
// This turns the per-window K-rope into one global projection pass.
//
// Everything between the f32 input and f32 output is bf16 and fits in the
// 192MB L2 (x:32MB, Q/K/V/O:32MB each), so direct global loads are cheap.
// All matmuls use v_wmma_f32_16x16x32_bf16 with f32 accumulation.
// ---------------------------------------------------------------------------

#define B_  8
#define T_  4096
#define C_  512
#define H_  8
#define D_  64
#define W_  128
#define NW_ (T_ / W_)          // 32 windows per batch
#define BT_ (B_ * T_)          // 32768 rows

typedef __attribute__((ext_vector_type(16))) __bf16 bf16x16;
typedef __attribute__((ext_vector_type(8)))  float  f32x8;

union AB { bf16x16 v; uint4 u[2]; };          // 32 bytes: one WMMA A/B operand
union V8 { __bf16 h[8]; uint4 u; };           // 16 bytes

static __device__ __forceinline__
f32x8 wmma_bf16(const AB& a, const AB& b, f32x8 c) {
    return __builtin_amdgcn_wmma_f32_16x16x32_bf16(
        false, a.v, false, b.v, (short)0, c, false, false);
}

// ---------------------------------------------------------------- prep ------
__global__ __launch_bounds__(256) void k_prep_x(
    const float* __restrict__ x, const unsigned char* __restrict__ mask,
    __bf16* __restrict__ xb)
{
    size_t i = (size_t)blockIdx.x * 256 + threadIdx.x;   // BT_*C_ elements
    float m = mask[i / C_] ? 1.0f : 0.0f;
    xb[i] = (__bf16)(x[i] * m);
}

__global__ __launch_bounds__(256) void k_prep_w(
    const float* __restrict__ w, __bf16* __restrict__ wt)
{
    int i = blockIdx.x * 256 + threadIdx.x;              // C_*C_ elements
    int n = i / C_, k = i % C_;
    wt[i] = (__bf16)w[(size_t)k * C_ + n];               // wt[n][k] = w[k][n]
}

// -------------------------------------------------- fused QKV projection ----
// grid (BT_/128, C_/128, 3), block 256 (8 waves, each 16 rows x 128 cols).
// z=0: Q (rope, row-major)  z=1: K (rope, row-major)  z=2: V (transposed).
__global__ __launch_bounds__(256) void k_proj(
    const __bf16* __restrict__ xb,
    const __bf16* __restrict__ Wqt, const __bf16* __restrict__ Wkt,
    const __bf16* __restrict__ Wvt,
    __bf16* __restrict__ Qrm, __bf16* __restrict__ Krm,
    __bf16* __restrict__ Vt)
{
    const int wave = threadIdx.x >> 5, lane = threadIdx.x & 31;
    const int row_base = blockIdx.x * 128 + wave * 16;
    const int col_base = blockIdx.y * 128;
    const int which    = blockIdx.z;
    const __bf16* Wt = (which == 0) ? Wqt : (which == 1) ? Wkt : Wvt;

    const int aoff = (lane & 16) ? 16 : 0;   // A: K-half select per lane group
    const int koff = (lane & 16) ? 16 : 0;   // B: K-half select per lane group
    const int arow = row_base + (lane & 15);
    const int rl   = (lane & 16) ? 8 : 0;    // C/D: row offset per lane group

    f32x8 acc[8] = {};
    for (int k = 0; k < C_; k += 32) {
        AB a;
        const __bf16* ap = xb + (size_t)arow * C_ + k;
        a.u[0] = *(const uint4*)(ap + aoff);
        a.u[1] = *(const uint4*)(ap + 8 + aoff);
#pragma unroll
        for (int n = 0; n < 8; ++n) {
            AB b;
            const __bf16* bp =
                Wt + (size_t)(col_base + n * 16 + (lane & 15)) * C_ + k + koff;
            b.u[0] = *(const uint4*)bp;
            b.u[1] = *(const uint4*)(bp + 8);
            acc[n] = wmma_bf16(a, b, acc[n]);
        }
    }

    if (which == 2) {
        // V: store transposed  Vt[((b*H+h)*D + d)][t]  (contiguous in t).
        const int bb = row_base / T_;
        const int t0 = (row_base % T_) + rl;
#pragma unroll
        for (int n = 0; n < 8; ++n) {
            int c = col_base + n * 16 + (lane & 15);
            int h = c / D_, d = c % D_;
            V8 tmp;
#pragma unroll
            for (int r = 0; r < 8; ++r) tmp.h[r] = (__bf16)acc[n][r];
            *(uint4*)(Vt + ((size_t)(bb * H_ + h) * D_ + d) * T_ + t0) = tmp.u;
        }
    } else {
        // Q/K: RoPE at absolute position t, store row-major [g][c].
        __bf16* dst = (which == 0) ? Qrm : Krm;
        const int tb = (row_base % T_) + rl;
#pragma unroll
        for (int hg = 0; hg < 2; ++hg) {          // two heads per 128 cols
#pragma unroll
            for (int j = 0; j < 2; ++j) {         // pair tiles (j, j+2): d, d+32
                f32x8 lo = acc[hg * 4 + j];
                f32x8 hi = acc[hg * 4 + j + 2];
                int dlo  = j * 16 + (lane & 15);             // in [0,32)
                float freq = __expf(-0.28782313662425572f * (float)dlo);
                int clo = col_base + hg * 64 + dlo;
#pragma unroll
                for (int r = 0; r < 8; ++r) {
                    float t = (float)(tb + r);
                    float sn, cs;
                    __sincosf(t * freq, &sn, &cs);
                    float olo = lo[r] * cs - hi[r] * sn;
                    float ohi = hi[r] * cs + lo[r] * sn;
                    size_t rowoff = (size_t)(row_base + rl + r) * C_;
                    dst[rowoff + clo]      = (__bf16)olo;
                    dst[rowoff + clo + 32] = (__bf16)ohi;
                }
            }
        }
    }
}

// ------------------------------------------------------- attention core -----
// grid (B_*NW_, H_, 2), block 128 (4 waves x 16 queries = 64 queries/WG).
// Dynamic LDS: 64*256 f32 scores (64KB) + 64*256 bf16 attn (32KB).
__global__ __launch_bounds__(128) void k_attn(
    const __bf16* __restrict__ Qrm, const __bf16* __restrict__ Krm,
    const __bf16* __restrict__ Vt,  __bf16* __restrict__ O)
{
    extern __shared__ char smem[];
    float*  sc  = (float*)smem;                  // [64][256]
    __bf16* att = (__bf16*)(smem + 64 * 256 * 4);// [64][256]

    const int wave = threadIdx.x >> 5, lane = threadIdx.x & 31;
    const int bb = blockIdx.x >> 5;              // batch
    const int w  = blockIdx.x & 31;              // window
    const int h  = blockIdx.y;
    const int half = blockIdx.z;                 // which 64-query half
    const int m0   = wave * 16;                  // query strip inside WG
    const int aoff = (lane & 16) ? 16 : 0;
    const int koff = (lane & 16) ? 16 : 0;
    const int rl   = (lane & 16) ? 8 : 0;
    const int keybase = w * W_ - 64;             // absolute t of key column 0

    // ---- load Q strip (16 queries x 64 d) as two A tiles --------------------
    const int tq = w * W_ + half * 64 + m0 + (lane & 15);
    const __bf16* qp = Qrm + (size_t)(bb * T_ + tq) * C_ + h * D_;
    AB aq0, aq1;
    aq0.u[0] = *(const uint4*)(qp + aoff);
    aq0.u[1] = *(const uint4*)(qp + 8 + aoff);
    aq1.u[0] = *(const uint4*)(qp + 32 + aoff);
    aq1.u[1] = *(const uint4*)(qp + 40 + aoff);

    // ---- scores = Q K^T, scaled + masked, into LDS --------------------------
#pragma unroll
    for (int nt = 0; nt < 16; ++nt) {
        int tk = keybase + nt * 16 + (lane & 15);    // this lane's key column
        bool valid = (tk >= 0) && (tk < T_);
        AB b0{}, b1{};
        if (valid) {
            const __bf16* kp =
                Krm + (size_t)(bb * T_ + tk) * C_ + h * D_ + koff;
            b0.u[0] = *(const uint4*)kp;
            b0.u[1] = *(const uint4*)(kp + 8);
            b1.u[0] = *(const uint4*)(kp + 32);
            b1.u[1] = *(const uint4*)(kp + 40);
        }
        f32x8 acc = {};
        acc = wmma_bf16(aq0, b0, acc);
        acc = wmma_bf16(aq1, b1, acc);
#pragma unroll
        for (int r = 0; r < 8; ++r) {
            float v = valid ? acc[r] * 0.125f : -1.0e9f;   // 1/sqrt(64)
            sc[(m0 + rl + r) * 256 + nt * 16 + (lane & 15)] = v;
        }
    }
    __syncthreads();

    // ---- softmax: one thread per query row ---------------------------------
    if (threadIdx.x < 64) {
        float* srow = sc + (size_t)threadIdx.x * 256;
        float mx = -3.0e38f;
        for (int k = 0; k < 256; ++k) mx = fmaxf(mx, srow[k]);
        float sum = 0.0f;
        for (int k = 0; k < 256; ++k) {
            float e = __expf(srow[k] - mx);
            srow[k] = e; sum += e;
        }
        float rinv = 1.0f / sum;
        __bf16* arow = att + (size_t)threadIdx.x * 256;
        for (int k = 0; k < 256; ++k) arow[k] = (__bf16)(srow[k] * rinv);
    }
    __syncthreads();

    // ---- out = attn @ V  (K-dim = 256 keys, N = 64 d) -----------------------
    AB at[8];
    const __bf16* ap = att + (size_t)(m0 + (lane & 15)) * 256;
#pragma unroll
    for (int kk = 0; kk < 8; ++kk) {
        at[kk].u[0] = *(const uint4*)(ap + kk * 32 + aoff);
        at[kk].u[1] = *(const uint4*)(ap + kk * 32 + 8 + aoff);
    }
    f32x8 oacc[4] = {};
#pragma unroll
    for (int nt = 0; nt < 4; ++nt) {
        int d = nt * 16 + (lane & 15);
        const __bf16* vp = Vt + ((size_t)(bb * H_ + h) * D_ + d) * T_;
#pragma unroll
        for (int kk = 0; kk < 8; ++kk) {
            int c0 = keybase + kk * 32 + koff;
            // 16-chunks never straddle the t=0 / t=T edges; clamp keeps loads
            // in-bounds, and OOB keys carry exactly-zero attn weight.
            int c0c = c0 < 0 ? 0 : (c0 > T_ - 16 ? T_ - 16 : c0);
            AB bv;
            bv.u[0] = *(const uint4*)(vp + c0c);
            bv.u[1] = *(const uint4*)(vp + c0c + 8);
            oacc[nt] = wmma_bf16(at[kk], bv, oacc[nt]);
        }
    }
    const int g0 = bb * T_ + w * W_ + half * 64 + m0;
#pragma unroll
    for (int nt = 0; nt < 4; ++nt) {
        int c = h * D_ + nt * 16 + (lane & 15);
#pragma unroll
        for (int r = 0; r < 8; ++r)
            O[(size_t)(g0 + rl + r) * C_ + c] = (__bf16)oacc[nt][r];
    }
}

// ------------------------------------------------------ output projection ---
// grid (BT_/128, C_/128), block 256. out = O @ Wo (f32 result).
__global__ __launch_bounds__(256) void k_outproj(
    const __bf16* __restrict__ O, const __bf16* __restrict__ Wot,
    float* __restrict__ out)
{
    const int wave = threadIdx.x >> 5, lane = threadIdx.x & 31;
    const int row_base = blockIdx.x * 128 + wave * 16;
    const int col_base = blockIdx.y * 128;
    const int aoff = (lane & 16) ? 16 : 0;
    const int koff = (lane & 16) ? 16 : 0;
    const int rl   = (lane & 16) ? 8 : 0;
    const int arow = row_base + (lane & 15);

    f32x8 acc[8] = {};
    for (int k = 0; k < C_; k += 32) {
        AB a;
        const __bf16* ap = O + (size_t)arow * C_ + k;
        a.u[0] = *(const uint4*)(ap + aoff);
        a.u[1] = *(const uint4*)(ap + 8 + aoff);
#pragma unroll
        for (int n = 0; n < 8; ++n) {
            AB b;
            const __bf16* bp =
                Wot + (size_t)(col_base + n * 16 + (lane & 15)) * C_ + k + koff;
            b.u[0] = *(const uint4*)bp;
            b.u[1] = *(const uint4*)(bp + 8);
            acc[n] = wmma_bf16(a, b, acc[n]);
        }
    }
#pragma unroll
    for (int n = 0; n < 8; ++n) {
        int c = col_base + n * 16 + (lane & 15);
#pragma unroll
        for (int r = 0; r < 8; ++r)
            out[(size_t)(row_base + rl + r) * C_ + c] = acc[n][r];
    }
}

// ---------------------------------------------------------------------------
extern "C" void kernel_launch(void* const* d_in, const int* in_sizes, int n_in,
                              void* d_out, int out_size, void* d_ws, size_t ws_size,
                              hipStream_t stream) {
    (void)in_sizes; (void)n_in; (void)out_size; (void)ws_size;
    const float*         x    = (const float*)d_in[0];
    const unsigned char* mask = (const unsigned char*)d_in[1];
    const float*         Wq   = (const float*)d_in[2];
    const float*         Wk   = (const float*)d_in[3];
    const float*         Wv   = (const float*)d_in[4];
    const float*         Wo   = (const float*)d_in[5];
    float*               out  = (float*)d_out;

    // Workspace carve-up (all bf16): ~162 MB total.
    char* p = (char*)d_ws;
    __bf16* xb  = (__bf16*)p; p += (size_t)BT_ * C_ * 2;
    __bf16* Wqt = (__bf16*)p; p += (size_t)C_ * C_ * 2;
    __bf16* Wkt = (__bf16*)p; p += (size_t)C_ * C_ * 2;
    __bf16* Wvt = (__bf16*)p; p += (size_t)C_ * C_ * 2;
    __bf16* Wot = (__bf16*)p; p += (size_t)C_ * C_ * 2;
    __bf16* Qrm = (__bf16*)p; p += (size_t)BT_ * C_ * 2;
    __bf16* Krm = (__bf16*)p; p += (size_t)BT_ * C_ * 2;
    __bf16* Vt  = (__bf16*)p; p += (size_t)BT_ * C_ * 2;
    __bf16* Obf = (__bf16*)p; p += (size_t)BT_ * C_ * 2;

    k_prep_x<<<(BT_ * C_) / 256, 256, 0, stream>>>(x, mask, xb);
    k_prep_w<<<(C_ * C_) / 256, 256, 0, stream>>>(Wq, Wqt);
    k_prep_w<<<(C_ * C_) / 256, 256, 0, stream>>>(Wk, Wkt);
    k_prep_w<<<(C_ * C_) / 256, 256, 0, stream>>>(Wv, Wvt);
    k_prep_w<<<(C_ * C_) / 256, 256, 0, stream>>>(Wo, Wot);

    k_proj<<<dim3(BT_ / 128, C_ / 128, 3), 256, 0, stream>>>(
        xb, Wqt, Wkt, Wvt, Qrm, Krm, Vt);

    k_attn<<<dim3(B_ * NW_, H_, 2), 128, 64 * 256 * 4 + 64 * 256 * 2, stream>>>(
        Qrm, Krm, Vt, Obf);

    k_outproj<<<dim3(BT_ / 128, C_ / 128), 256, 0, stream>>>(Obf, Wot, out);
}
